// GraphNetBlock_44143673869053
// MI455X (gfx1250) — matrix operand order repaired and checked by reading, not compiled, not run on hardware
//
#include <hip/hip_runtime.h>

typedef __attribute__((ext_vector_type(16))) __bf16 v16bf;
typedef __attribute__((ext_vector_type(8)))  float  v8f;

#define N_NODES 100000
#define N_EDGES 500000
#define BN_EPS  1e-5f

union Frag {
  v16bf v;
  __bf16 e[16];
  uint4 q[2];
};

// ---- fragment helpers -------------------------------------------------------

// A fragment (16x32 bf16) from fp32 row: elements 0..7 = feats k0..k0+7,
// elements 8..15 = feats k0+16..k0+23 (k0 already includes the lane-half shift).
__device__ __forceinline__ v16bf afrag_f32(const float* __restrict__ p, int k0) {
  float4 x0 = *(const float4*)(p + k0);
  float4 x1 = *(const float4*)(p + k0 + 4);
  float4 x2 = *(const float4*)(p + k0 + 16);
  float4 x3 = *(const float4*)(p + k0 + 20);
  Frag u;
  u.e[0] = (__bf16)x0.x; u.e[1] = (__bf16)x0.y; u.e[2]  = (__bf16)x0.z; u.e[3]  = (__bf16)x0.w;
  u.e[4] = (__bf16)x1.x; u.e[5] = (__bf16)x1.y; u.e[6]  = (__bf16)x1.z; u.e[7]  = (__bf16)x1.w;
  u.e[8] = (__bf16)x2.x; u.e[9] = (__bf16)x2.y; u.e[10] = (__bf16)x2.z; u.e[11] = (__bf16)x2.w;
  u.e[12] = (__bf16)x3.x; u.e[13] = (__bf16)x3.y; u.e[14] = (__bf16)x3.z; u.e[15] = (__bf16)x3.w;
  return u.v;
}

// A fragment from contiguous bf16 row (LDS h1 staging), 2x 16B loads.
__device__ __forceinline__ v16bf afrag_bf16(const __bf16* row, int k0) {
  Frag u;
  u.q[0] = *(const uint4*)(row + k0);
  u.q[1] = *(const uint4*)(row + k0 + 16);
  return u.v;
}

// B fragment from pre-swizzled weight storage: frag fid, each lane reads its
// own contiguous 32B (two b128 loads).
__device__ __forceinline__ v16bf bfrag_sw(const __bf16* base, int fid, int lane) {
  Frag u;
  const uint4* p = (const uint4*)(base + ((size_t)fid << 9) + (lane << 4));
  u.q[0] = p[0];
  u.q[1] = p[1];
  return u.v;
}

__device__ __forceinline__ v8f splat8(float x) {
  v8f v = {x, x, x, x, x, x, x, x};
  return v;
}
__device__ __forceinline__ float elu(float x) {
  return x > 0.f ? x : (__expf(x) - 1.f);
}

// ---- kernel 0: weight convert + swizzle to bf16 B-fragment layout, zero bufs
__global__ __launch_bounds__(256)
void prep_kernel(const float* __restrict__ ew1, const float* __restrict__ ew2,
                 const float* __restrict__ nw1, const float* __restrict__ nw2,
                 __bf16* __restrict__ W1S, __bf16* __restrict__ W2S,
                 __bf16* __restrict__ NW1S, __bf16* __restrict__ NW2S,
                 float* __restrict__ agg, float* __restrict__ statsE,
                 float* __restrict__ statsN) {
  const int stride = gridDim.x * blockDim.x;
  const int g = blockIdx.x * blockDim.x + threadIdx.x;
  for (int i = g; i < N_NODES * 32; i += stride) agg[i] = 0.f;
  if (g < 64)  statsE[g] = 0.f;
  if (g < 256) statsN[g] = 0.f;
  // edge W1: [128 out][288 in] -> 72 frags (c<9, t<8)
  for (int idx = g; idx < 72 * 512; idx += stride) {
    int fid = idx >> 9, rem = idx & 511, ln = rem >> 4, e = rem & 15;
    int c = fid >> 3, t = fid & 7;
    int o = t * 16 + (ln & 15);
    int i = c * 32 + ((ln >> 4) << 3) + (e & 7) + ((e >> 3) << 4);
    W1S[idx] = (__bf16)ew1[o * 288 + i];
  }
  // edge W2: [32 out][128 in] -> 8 frags (c<4, t<2)
  for (int idx = g; idx < 8 * 512; idx += stride) {
    int fid = idx >> 9, rem = idx & 511, ln = rem >> 4, e = rem & 15;
    int c = fid >> 1, t = fid & 1;
    int o = t * 16 + (ln & 15);
    int i = c * 32 + ((ln >> 4) << 3) + (e & 7) + ((e >> 3) << 4);
    W2S[idx] = (__bf16)ew2[o * 128 + i];
  }
  // node W1: [256 out][160 in] -> 80 frags (c<5, t<16)
  for (int idx = g; idx < 80 * 512; idx += stride) {
    int fid = idx >> 9, rem = idx & 511, ln = rem >> 4, e = rem & 15;
    int c = fid >> 4, t = fid & 15;
    int o = t * 16 + (ln & 15);
    int i = c * 32 + ((ln >> 4) << 3) + (e & 7) + ((e >> 3) << 4);
    NW1S[idx] = (__bf16)nw1[o * 160 + i];
  }
  // node W2: [128 out][256 in] -> 64 frags (c<8, t<8)
  for (int idx = g; idx < 64 * 512; idx += stride) {
    int fid = idx >> 9, rem = idx & 511, ln = rem >> 4, e = rem & 15;
    int c = fid >> 3, t = fid & 7;
    int o = t * 16 + (ln & 15);
    int i = c * 32 + ((ln >> 4) << 3) + (e & 7) + ((e >> 3) << 4);
    NW2S[idx] = (__bf16)nw2[o * 256 + i];
  }
}

// ---- kernel 1: edge MLP (WMMA bf16, f32 accum) ------------------------------
__global__ __launch_bounds__(256)
void edge_mlp_kernel(const float* __restrict__ nodes, const float* __restrict__ edges,
                     const int* __restrict__ pair, const float* __restrict__ eb1,
                     const float* __restrict__ eb2, const __bf16* __restrict__ W1S,
                     const __bf16* __restrict__ W2S, float* __restrict__ h2e,
                     float* __restrict__ statsE) {
  __shared__ __align__(16) __bf16 sW1[72 * 512];     // 73728 B, swizzled frags
  __shared__ __align__(16) __bf16 sW2[8 * 512];      // 8192 B
  __shared__ __align__(16) __bf16 sH1[8][16 * 136];  // per-wave h1, padded

  {
    const uint4* s1 = (const uint4*)W1S;
    uint4* d1 = (uint4*)sW1;
    for (int i = threadIdx.x; i < 4608; i += 256) d1[i] = s1[i];
    const uint4* s2 = (const uint4*)W2S;
    uint4* d2 = (uint4*)sW2;
    for (int i = threadIdx.x; i < 512; i += 256) d2[i] = s2[i];
  }
  __syncthreads();

  const int lane = threadIdx.x & 31;
  const int wid = threadIdx.x >> 5;
  const int m = lane & 15;
  const int half8 = (lane >> 4) << 3;  // 0 or 8
  __bf16* h1w = &sH1[wid][0];
  const __bf16* h1row = &sH1[wid][m * 136];

  float sSum[2] = {0.f, 0.f}, sSq[2] = {0.f, 0.f};
  const int NT = N_EDGES / 16;

  for (int tile = blockIdx.x * 8 + wid; tile < NT; tile += gridDim.x * 8) {
    const int e = tile * 16 + m;
    __builtin_prefetch(&pair[2 * (e + (int)gridDim.x * 128)], 0, 1);
    const int s = pair[2 * e + 0];
    const int r = pair[2 * e + 1];
    const float* ps = nodes + (size_t)s * 128;
    const float* pr = nodes + (size_t)r * 128;
    const float* pe = edges + (size_t)e * 32;

    v16bf A[9];
#pragma unroll
    for (int c = 0; c < 4; ++c) A[c] = afrag_f32(ps, c * 32 + half8);
#pragma unroll
    for (int c = 0; c < 4; ++c) A[4 + c] = afrag_f32(pr, c * 32 + half8);
    A[8] = afrag_f32(pe, half8);

    // layer 1: 288 -> 128, ELU, stage bf16 h1 in LDS
#pragma unroll
    for (int t = 0; t < 8; ++t) {
      v8f acc = splat8(eb1[t * 16 + m]);
#pragma unroll
      for (int c = 0; c < 9; ++c)
        acc = __builtin_amdgcn_wmma_f32_16x16x32_bf16(
            false, A[c], false, bfrag_sw(sW1, c * 8 + t, lane), (short)0, acc,
            false, false);
#pragma unroll
      for (int rr = 0; rr < 8; ++rr)
        h1w[(half8 + rr) * 136 + t * 16 + m] = (__bf16)elu(acc[rr]);
    }
    asm volatile("s_wait_dscnt 0" ::: "memory");  // h1 stores -> A2 loads (intra-wave)

    v16bf A2[4];
#pragma unroll
    for (int c = 0; c < 4; ++c) A2[c] = afrag_bf16(h1row, c * 32 + half8);

    // layer 2: 128 -> 32, ELU, pre-BN h2 to workspace + stats
#pragma unroll
    for (int t = 0; t < 2; ++t) {
      v8f acc = splat8(eb2[t * 16 + m]);
#pragma unroll
      for (int c = 0; c < 4; ++c)
        acc = __builtin_amdgcn_wmma_f32_16x16x32_bf16(
            false, A2[c], false, bfrag_sw(sW2, c * 2 + t, lane), (short)0, acc,
            false, false);
      float* outp = h2e + (size_t)(tile * 16) * 32 + t * 16 + m;
#pragma unroll
      for (int rr = 0; rr < 8; ++rr) {
        float y = elu(acc[rr]);
        outp[(size_t)(half8 + rr) * 32] = y;
        sSum[t] += y;
        sSq[t] += y * y;
      }
    }
  }
#pragma unroll
  for (int t = 0; t < 2; ++t) {
    unsafeAtomicAdd(&statsE[t * 16 + m], sSum[t]);
    unsafeAtomicAdd(&statsE[32 + t * 16 + m], sSq[t]);
  }
}

// ---- kernel 2: edge BN apply + output + scatter-add into agg ----------------
__global__ __launch_bounds__(256)
void edge_bn_scatter_kernel(const float* __restrict__ h2e, const float* __restrict__ statsE,
                            const float* __restrict__ eg, const float* __restrict__ ebt,
                            const float* __restrict__ edges, const int* __restrict__ pair,
                            float* __restrict__ agg, float* __restrict__ outE) {
  const size_t tid = (size_t)blockIdx.x * 256 + threadIdx.x;  // exact: 16M threads
  const int f = (int)(tid & 31);
  const size_t e = tid >> 5;
  const float inv = 1.f / (float)N_EDGES;
  const float mu = statsE[f] * inv;
  const float var = statsE[32 + f] * inv - mu * mu;
  const float sc = eg[f] * rsqrtf(var + BN_EPS);
  const float sh = ebt[f] - mu * sc;
  const float val = h2e[tid] * sc + sh;      // e0 = BN(h2)
  outE[tid] = val + edges[tid];              // new_edges + edges
  const int s = pair[2 * e + 0];
  const int r = pair[2 * e + 1];
  unsafeAtomicAdd(&agg[(size_t)s * 32 + f], val);
  unsafeAtomicAdd(&agg[(size_t)r * 32 + f], val);
}

// ---- kernel 3: node MLP (WMMA bf16) -----------------------------------------
__global__ __launch_bounds__(128)
void node_mlp_kernel(const float* __restrict__ nodes, const float* __restrict__ agg,
                     const float* __restrict__ nb1, const float* __restrict__ nb2,
                     const __bf16* __restrict__ NW1S, const __bf16* __restrict__ NW2S,
                     float* __restrict__ h2n, float* __restrict__ statsN) {
  __shared__ __align__(16) __bf16 sW1[80 * 512];     // 81920 B, swizzled frags
  __shared__ __align__(16) __bf16 sH1[4][16 * 264];  // per-wave h1, padded

  {
    const uint4* s1 = (const uint4*)NW1S;
    uint4* d1 = (uint4*)sW1;
    for (int i = threadIdx.x; i < 5120; i += 128) d1[i] = s1[i];
  }
  __syncthreads();

  const int lane = threadIdx.x & 31;
  const int wid = threadIdx.x >> 5;
  const int m = lane & 15;
  const int half8 = (lane >> 4) << 3;
  __bf16* h1w = &sH1[wid][0];
  const __bf16* h1row = &sH1[wid][m * 264];

  float sSum[8], sSq[8];
#pragma unroll
  for (int t = 0; t < 8; ++t) { sSum[t] = 0.f; sSq[t] = 0.f; }

  const int NT = N_NODES / 16;
  for (int tile = blockIdx.x * 4 + wid; tile < NT; tile += gridDim.x * 4) {
    const int n = tile * 16 + m;
    const float* pn = nodes + (size_t)n * 128;
    const float* pa = agg + (size_t)n * 32;

    v16bf A[5];
#pragma unroll
    for (int c = 0; c < 4; ++c) A[c] = afrag_f32(pn, c * 32 + half8);
    A[4] = afrag_f32(pa, half8);

    // layer 1: 160 -> 256, ELU -> LDS
#pragma unroll
    for (int t = 0; t < 16; ++t) {
      v8f acc = splat8(nb1[t * 16 + m]);
#pragma unroll
      for (int c = 0; c < 5; ++c)
        acc = __builtin_amdgcn_wmma_f32_16x16x32_bf16(
            false, A[c], false, bfrag_sw(sW1, c * 16 + t, lane), (short)0, acc,
            false, false);
#pragma unroll
      for (int rr = 0; rr < 8; ++rr)
        h1w[(half8 + rr) * 264 + t * 16 + m] = (__bf16)elu(acc[rr]);
    }
    asm volatile("s_wait_dscnt 0" ::: "memory");

    v16bf A2[8];
#pragma unroll
    for (int c = 0; c < 8; ++c) A2[c] = afrag_bf16(h1row, c * 32 + half8);

    // layer 2: 256 -> 128 (B frags streamed from L2-resident swizzled global)
#pragma unroll
    for (int t = 0; t < 8; ++t) {
      v8f acc = splat8(nb2[t * 16 + m]);
#pragma unroll
      for (int c = 0; c < 8; ++c)
        acc = __builtin_amdgcn_wmma_f32_16x16x32_bf16(
            false, A2[c], false, bfrag_sw(NW2S, c * 8 + t, lane), (short)0, acc,
            false, false);
      float* outp = h2n + (size_t)(tile * 16) * 128 + t * 16 + m;
#pragma unroll
      for (int rr = 0; rr < 8; ++rr) {
        float y = elu(acc[rr]);
        outp[(size_t)(half8 + rr) * 128] = y;
        sSum[t] += y;
        sSq[t] += y * y;
      }
    }
  }
#pragma unroll
  for (int t = 0; t < 8; ++t) {
    unsafeAtomicAdd(&statsN[t * 16 + m], sSum[t]);
    unsafeAtomicAdd(&statsN[128 + t * 16 + m], sSq[t]);
  }
}

// ---- kernel 4: node BN + residual -------------------------------------------
__global__ __launch_bounds__(256)
void node_bn_kernel(const float* __restrict__ h2n, const float* __restrict__ statsN,
                    const float* __restrict__ ng, const float* __restrict__ nbt,
                    const float* __restrict__ nodes, float* __restrict__ outN) {
  const size_t tid = (size_t)blockIdx.x * 256 + threadIdx.x;  // exact: 12.8M threads
  const int f = (int)(tid & 127);
  const float inv = 1.f / (float)N_NODES;
  const float mu = statsN[f] * inv;
  const float var = statsN[128 + f] * inv - mu * mu;
  const float sc = ng[f] * rsqrtf(var + BN_EPS);
  const float sh = nbt[f] - mu * sc;
  outN[tid] = h2n[tid] * sc + sh + nodes[tid];
}

// ---- host launch ------------------------------------------------------------
extern "C" void kernel_launch(void* const* d_in, const int* in_sizes, int n_in,
                              void* d_out, int out_size, void* d_ws, size_t ws_size,
                              hipStream_t stream) {
  const float* nodes = (const float*)d_in[0];
  const float* edges = (const float*)d_in[1];
  const int* pair = (const int*)d_in[2];
  const float* ew1 = (const float*)d_in[3];
  const float* eb1 = (const float*)d_in[4];
  const float* ew2 = (const float*)d_in[5];
  const float* eb2 = (const float*)d_in[6];
  const float* eg = (const float*)d_in[7];
  const float* ebt = (const float*)d_in[8];
  const float* nw1 = (const float*)d_in[9];
  const float* nb1 = (const float*)d_in[10];
  const float* nw2 = (const float*)d_in[11];
  const float* nb2 = (const float*)d_in[12];
  const float* ng = (const float*)d_in[13];
  const float* nbt = (const float*)d_in[14];

  char* ws = (char*)d_ws;
  __bf16* W1S  = (__bf16*)(ws + 0);          //  73728 B
  __bf16* W2S  = (__bf16*)(ws + 73728);      //   8192 B
  __bf16* NW1S = (__bf16*)(ws + 81920);      //  81920 B
  __bf16* NW2S = (__bf16*)(ws + 163840);     //  65536 B
  float* statsE = (float*)(ws + 229376);     //    256 B (32 sum + 32 sumsq)
  float* statsN = (float*)(ws + 229632);     //   1024 B (128 sum + 128 sumsq)
  float* agg    = (float*)(ws + 230656);     // 12.8 MB
  float* h2e    = (float*)(ws + 13030656);   // 64.0 MB
  float* h2n    = (float*)(ws + 77030656);   // 51.2 MB  (total ~128.2 MB)

  float* outN = (float*)d_out;                       // [100000,128]
  float* outE = outN + (size_t)N_NODES * 128;        // [500000,32]

  prep_kernel<<<2048, 256, 0, stream>>>(ew1, ew2, nw1, nw2, W1S, W2S, NW1S,
                                        NW2S, agg, statsE, statsN);
  edge_mlp_kernel<<<1024, 256, 0, stream>>>(nodes, edges, pair, eb1, eb2, W1S,
                                            W2S, h2e, statsE);
  edge_bn_scatter_kernel<<<62500, 256, 0, stream>>>(h2e, statsE, eg, ebt, edges,
                                                    pair, agg, outE);
  node_mlp_kernel<<<1024, 128, 0, stream>>>(nodes, agg, nb1, nb2, NW1S, NW2S,
                                            h2n, statsN);
  node_bn_kernel<<<50000, 256, 0, stream>>>(h2n, statsN, ng, nbt, nodes, outN);
}
